// AlignFeatureLayer_11862699671621
// MI455X (gfx1250) — matrix operand order repaired and checked by reading, not compiled, not run on hardware
//
#include <hip/hip_runtime.h>
#include <math.h>

#define B_  32
#define LD_ 2048
#define LQ_ 512
#define E_  768

typedef float  v8f  __attribute__((ext_vector_type(8)));
typedef __bf16 v4bf __attribute__((ext_vector_type(4)));
typedef __bf16 v8bf __attribute__((ext_vector_type(8)));
typedef __bf16 v16bf __attribute__((ext_vector_type(16)));

// bf16x3 building block: D = A(16x32 bf16) x B(32x16 bf16) + C(16x16 f32)
__device__ __forceinline__ v8f wmma_bf16(v16bf a, v16bf b, v8f c) {
  return __builtin_amdgcn_wmma_f32_16x16x32_bf16(false, a, false, b, (short)0, c,
                                                 false, false);
}

// v16bf fragment = two contiguous 16B chunks (ISA 7.12.2 16-bit layouts:
//  A: lane<16 -> K {k0..k0+7, k0+16..k0+23}; lane>=16 -> +8.  B: same K order as A)
__device__ __forceinline__ v16bf ldfrag(const __bf16* p0, const __bf16* p1) {
  const v8bf a = *(const v8bf*)p0;
  const v8bf b = *(const v8bf*)p1;
  return __builtin_shufflevector(a, b, 0, 1, 2, 3, 4, 5, 6, 7,
                                 8, 9, 10, 11, 12, 13, 14, 15);
}

struct bf2 { __bf16 h, l; };
__device__ __forceinline__ bf2 bsplit(float x) {
  bf2 r;
  r.h = (__bf16)x;
  r.l = (__bf16)(x - (float)r.h);  // exact residual in f32; ~16 mantissa bits total
  return r;
}

#define ASTR  776   // bf16 stride for 768-wide tiles: bank = (4*row + k/2) % 64
#define ALSTR 520   // bf16 stride for 512-wide alpha
#define SSTR  516   // f32 stride for scores
#define BSTR  40    // bf16 stride for staged [128 x 32] B tiles: banks 20*row -> disjoint

// ---------------------------------------------------------------------------
// Setup kernel 1: split W -> bf16 hi/lo planes (row-major == WMMA B layout)
// ---------------------------------------------------------------------------
__global__ __launch_bounds__(256) void convert_w_kernel(
    const float* __restrict__ W, __bf16* __restrict__ wh, __bf16* __restrict__ wl) {
  const int i = blockIdx.x * 256 + threadIdx.x;   // grid covers 768*768 exactly
  const bf2 s = bsplit(W[i]);
  wh[i] = s.h; wl[i] = s.l;
}

// ---------------------------------------------------------------------------
// Setup kernel 2: q_emb [b][m][e] -> transposed bf16 hi/lo qet [b][e][m]
// ---------------------------------------------------------------------------
__global__ __launch_bounds__(256) void qet_kernel(
    const float* __restrict__ q_emb, __bf16* __restrict__ qeth,
    __bf16* __restrict__ qetl) {
  __shared__ float tile[32][33];
  const int b = blockIdx.z, e0 = blockIdx.x * 32, m0 = blockIdx.y * 32;
  const int x = threadIdx.x & 31, y = threadIdx.x >> 5;   // y in 0..7
#pragma unroll
  for (int j = 0; j < 4; ++j)
    tile[y + 8 * j][x] = q_emb[(b * LQ_ + m0 + y + 8 * j) * E_ + e0 + x];
  __syncthreads();
#pragma unroll
  for (int j = 0; j < 4; ++j) {
    const bf2 s = bsplit(tile[x][y + 8 * j]);
    const int o = (b * E_ + e0 + y + 8 * j) * LQ_ + m0 + x;
    qeth[o] = s.h; qetl[o] = s.l;
  }
}

// ---------------------------------------------------------------------------
// Kernel 3: q_proj = relu(q_emb @ W^T + b) -> bf16 hi/lo row-major [b][m][e]
// Per block: 16 m-rows. A tile resident in LDS (split once); B staged K=32/round.
// dyn LDS: A 49,664 + B 20,480 = 70,144 B
// ---------------------------------------------------------------------------
__global__ __launch_bounds__(256) void qproj_kernel(
    const float* __restrict__ q_emb, const __bf16* __restrict__ wh,
    const __bf16* __restrict__ wl, const float* __restrict__ bias,
    __bf16* __restrict__ qph, __bf16* __restrict__ qpl) {
  extern __shared__ __align__(16) unsigned char smem[];
  __bf16* sAh = (__bf16*)smem;                 // [16][ASTR]
  __bf16* sAl = sAh + 16 * ASTR;
  __bf16* sBh = (__bf16*)(smem + 49664);       // [128][BSTR]
  __bf16* sBl = sBh + 128 * BSTR;

  const int b = blockIdx.y, m0 = blockIdx.x * 16;
  const int tid = threadIdx.x, wave = tid >> 5, lane = tid & 31;
  const int lanelo = lane & 15, hi = lane >> 4;

  // stage + split A tile [16][768]
  {
    const int r = tid >> 4, cb = (tid & 15) * 4;
#pragma unroll
    for (int j = 0; j < 12; ++j) {
      const int c = cb + 64 * j;
      const float4 v = *(const float4*)&q_emb[(b * LQ_ + m0 + r) * E_ + c];
      const bf2 s0 = bsplit(v.x), s1 = bsplit(v.y);
      const bf2 s2 = bsplit(v.z), s3 = bsplit(v.w);
      v4bf h, l;
      h[0] = s0.h; h[1] = s1.h; h[2] = s2.h; h[3] = s3.h;
      l[0] = s0.l; l[1] = s1.l; l[2] = s2.l; l[3] = s3.l;
      *(v4bf*)&sAh[r * ASTR + c] = h;
      *(v4bf*)&sAl[r * ASTR + c] = l;
    }
  }
  __syncthreads();

  const int br = tid >> 1, bc = (tid & 1) * 16;   // B-stage coords
  for (int np = 0; np < 6; ++np) {
    const int n0 = np * 128;
    v8f acc = {0.f, 0.f, 0.f, 0.f, 0.f, 0.f, 0.f, 0.f};
    for (int kc = 0; kc < 24; ++kc) {
      const int k0 = kc * 32;
      __syncthreads();
      const int g = (n0 + br) * E_ + k0 + bc;
      const v8bf w0 = *(const v8bf*)&wh[g];
      const v8bf w1 = *(const v8bf*)&wh[g + 8];
      const v8bf w2 = *(const v8bf*)&wl[g];
      const v8bf w3 = *(const v8bf*)&wl[g + 8];
      *(v8bf*)&sBh[br * BSTR + bc]     = w0;
      *(v8bf*)&sBh[br * BSTR + bc + 8] = w1;
      *(v8bf*)&sBl[br * BSTR + bc]     = w2;
      *(v8bf*)&sBl[br * BSTR + bc + 8] = w3;
      __syncthreads();
      const int ka = lanelo * ASTR + k0 + 8 * hi;
      const v16bf Ah = ldfrag(&sAh[ka], &sAh[ka + 16]);
      const v16bf Al = ldfrag(&sAl[ka], &sAl[ka + 16]);
      const int kb = (wave * 16 + lanelo) * BSTR + 8 * hi;
      const v16bf Bh = ldfrag(&sBh[kb], &sBh[kb + 16]);
      const v16bf Bl = ldfrag(&sBl[kb], &sBl[kb + 16]);
      acc = wmma_bf16(Ah, Bh, acc);
      acc = wmma_bf16(Ah, Bl, acc);
      acc = wmma_bf16(Al, Bh, acc);
    }
    const int e = n0 + wave * 16 + lanelo;
    const float bv = bias[e];
#pragma unroll
    for (int r = 0; r < 8; ++r) {
      const int row = r + 8 * hi;
      const bf2 s = bsplit(fmaxf(acc[r] + bv, 0.0f));
      const int o = (b * LQ_ + m0 + row) * E_ + e;
      qph[o] = s.h; qpl[o] = s.l;
    }
  }
}

// ---------------------------------------------------------------------------
// Kernel 4 (fused attention), per block = (batch, 16 d-rows), bf16x3 everywhere.
// dyn LDS regions: R1 0..49,663   : d_emb hi/lo [16][ASTR]  -> later scores f32 [16][SSTR]
//                  R2 49,664..    : d_proj hi/lo [16][ASTR] -> later alpha hi/lo [16][ALSTR]
//                  R3 99,328..    : staged B hi/lo [128][BSTR]      total 119,808 B
// ---------------------------------------------------------------------------
__global__ __launch_bounds__(256) void attn_kernel(
    const float* __restrict__ d_emb, const float* __restrict__ q_emb,
    const unsigned char* __restrict__ qmask, const float* __restrict__ bias,
    const __bf16* __restrict__ wh, const __bf16* __restrict__ wl,
    const __bf16* __restrict__ qph, const __bf16* __restrict__ qpl,
    const __bf16* __restrict__ qeth, const __bf16* __restrict__ qetl,
    float* __restrict__ out_align, float* __restrict__ out_alpha) {
  extern __shared__ __align__(16) unsigned char smem[];
  __bf16* sAh  = (__bf16*)smem;                  // R1: d_emb tile
  __bf16* sAl  = sAh + 16 * ASTR;
  float*  sSc  = (float*)smem;                   // R1 alias: scores [16][SSTR]
  __bf16* sDPh = (__bf16*)(smem + 49664);        // R2: d_proj tile
  __bf16* sDPl = sDPh + 16 * ASTR;
  __bf16* sALh = (__bf16*)(smem + 49664);        // R2 alias: alpha hi/lo
  __bf16* sALl = sALh + 16 * ALSTR;
  __bf16* sBh  = (__bf16*)(smem + 99328);        // R3: staged B
  __bf16* sBl  = sBh + 128 * BSTR;

  const int b = blockIdx.y, l0 = blockIdx.x * 16;
  const int tid = threadIdx.x, wave = tid >> 5, lane = tid & 31;
  const int lanelo = lane & 15, hi = lane >> 4;
  const int br = tid >> 1, bc = (tid & 1) * 16;
  const float NEGINF = -__builtin_inff();

  // ---- stage + split d_emb tile ----
  {
    const int r = tid >> 4, cb = (tid & 15) * 4;
#pragma unroll
    for (int j = 0; j < 12; ++j) {
      const int c = cb + 64 * j;
      const float4 v = *(const float4*)&d_emb[(b * LD_ + l0 + r) * E_ + c];
      const bf2 s0 = bsplit(v.x), s1 = bsplit(v.y);
      const bf2 s2 = bsplit(v.z), s3 = bsplit(v.w);
      v4bf h, l;
      h[0] = s0.h; h[1] = s1.h; h[2] = s2.h; h[3] = s3.h;
      l[0] = s0.l; l[1] = s1.l; l[2] = s2.l; l[3] = s3.l;
      *(v4bf*)&sAh[r * ASTR + c] = h;
      *(v4bf*)&sAl[r * ASTR + c] = l;
    }
  }
  __syncthreads();

  // ---- phase 1: d_proj = relu(d_emb @ W^T + b) -> R2 (bf16 hi/lo) ----
  for (int np = 0; np < 6; ++np) {
    const int n0 = np * 128;
    v8f acc = {0.f, 0.f, 0.f, 0.f, 0.f, 0.f, 0.f, 0.f};
    for (int kc = 0; kc < 24; ++kc) {
      const int k0 = kc * 32;
      __syncthreads();
      const int g = (n0 + br) * E_ + k0 + bc;
      const v8bf w0 = *(const v8bf*)&wh[g];
      const v8bf w1 = *(const v8bf*)&wh[g + 8];
      const v8bf w2 = *(const v8bf*)&wl[g];
      const v8bf w3 = *(const v8bf*)&wl[g + 8];
      *(v8bf*)&sBh[br * BSTR + bc]     = w0;
      *(v8bf*)&sBh[br * BSTR + bc + 8] = w1;
      *(v8bf*)&sBl[br * BSTR + bc]     = w2;
      *(v8bf*)&sBl[br * BSTR + bc + 8] = w3;
      __syncthreads();
      const int ka = lanelo * ASTR + k0 + 8 * hi;
      const v16bf Ah = ldfrag(&sAh[ka], &sAh[ka + 16]);
      const v16bf Al = ldfrag(&sAl[ka], &sAl[ka + 16]);
      const int kb = (wave * 16 + lanelo) * BSTR + 8 * hi;
      const v16bf Bh = ldfrag(&sBh[kb], &sBh[kb + 16]);
      const v16bf Bl = ldfrag(&sBl[kb], &sBl[kb + 16]);
      acc = wmma_bf16(Ah, Bh, acc);
      acc = wmma_bf16(Ah, Bl, acc);
      acc = wmma_bf16(Al, Bh, acc);
    }
    const int e = n0 + wave * 16 + lanelo;
    const float bv = bias[e];
#pragma unroll
    for (int r = 0; r < 8; ++r) {
      const int row = r + 8 * hi;
      const bf2 s = bsplit(fmaxf(acc[r] + bv, 0.0f));
      sDPh[row * ASTR + e] = s.h;
      sDPl[row * ASTR + e] = s.l;
    }
  }

  // ---- phase 2: scores = d_proj @ q_proj^T -> R1 (f32, masked) ----
  for (int p = 0; p < 4; ++p) {
    const int m0 = p * 128;
    v8f acc = {0.f, 0.f, 0.f, 0.f, 0.f, 0.f, 0.f, 0.f};
    for (int kc = 0; kc < 24; ++kc) {
      const int k0 = kc * 32;
      __syncthreads();
      const int g = (b * LQ_ + m0 + br) * E_ + k0 + bc;
      const v8bf w0 = *(const v8bf*)&qph[g];
      const v8bf w1 = *(const v8bf*)&qph[g + 8];
      const v8bf w2 = *(const v8bf*)&qpl[g];
      const v8bf w3 = *(const v8bf*)&qpl[g + 8];
      *(v8bf*)&sBh[br * BSTR + bc]     = w0;
      *(v8bf*)&sBh[br * BSTR + bc + 8] = w1;
      *(v8bf*)&sBl[br * BSTR + bc]     = w2;
      *(v8bf*)&sBl[br * BSTR + bc + 8] = w3;
      __syncthreads();
      const int ka = lanelo * ASTR + k0 + 8 * hi;
      const v16bf Ah = ldfrag(&sDPh[ka], &sDPh[ka + 16]);
      const v16bf Al = ldfrag(&sDPl[ka], &sDPl[ka + 16]);
      const int kb = (wave * 16 + lanelo) * BSTR + 8 * hi;
      const v16bf Bh = ldfrag(&sBh[kb], &sBh[kb + 16]);
      const v16bf Bl = ldfrag(&sBl[kb], &sBl[kb + 16]);
      acc = wmma_bf16(Ah, Bh, acc);
      acc = wmma_bf16(Ah, Bl, acc);
      acc = wmma_bf16(Al, Bh, acc);
    }
    const int m = m0 + wave * 16 + lanelo;
    const bool msk = qmask[b * LQ_ + m] != 0;
#pragma unroll
    for (int r = 0; r < 8; ++r)
      sSc[(r + 8 * hi) * SSTR + m] = msk ? NEGINF : acc[r];
  }
  __syncthreads();   // scores complete; d_proj (R2) dead

  // ---- phase 3: softmax per row (one 16-lane half-wave each); alpha -> R2 ----
  {
    const int row = wave * 2 + hi;
    float mx = NEGINF;
#pragma unroll 4
    for (int j = 0; j < 32; ++j)
      mx = fmaxf(mx, sSc[row * SSTR + j * 16 + lanelo]);
#pragma unroll
    for (int off = 8; off > 0; off >>= 1) mx = fmaxf(mx, __shfl_xor(mx, off, 16));
    float sum = 0.0f;
    float ev[32];
#pragma unroll 4
    for (int j = 0; j < 32; ++j) {
      ev[j] = __expf(sSc[row * SSTR + j * 16 + lanelo] - mx);
      sum += ev[j];
    }
#pragma unroll
    for (int off = 8; off > 0; off >>= 1) sum += __shfl_xor(sum, off, 16);
    const float inv = 1.0f / sum;
#pragma unroll 4
    for (int j = 0; j < 32; ++j) {
      const int col = j * 16 + lanelo;
      const float av = ev[j] * inv;
      out_alpha[(b * LD_ + l0 + row) * LQ_ + col] = av;
      const bf2 s = bsplit(av);
      sALh[row * ALSTR + col] = s.h;
      sALl[row * ALSTR + col] = s.l;
    }
  }
  __syncthreads();

  // ---- phase 4: align = alpha @ q_emb (B from transposed qet, staged) ----
  for (int np = 0; np < 6; ++np) {
    const int n0 = np * 128;
    v8f acc = {0.f, 0.f, 0.f, 0.f, 0.f, 0.f, 0.f, 0.f};
    for (int kc = 0; kc < 16; ++kc) {        // K = LQ = 512
      const int k0 = kc * 32;
      __syncthreads();
      const int g = (b * E_ + n0 + br) * LQ_ + k0 + bc;
      const v8bf w0 = *(const v8bf*)&qeth[g];
      const v8bf w1 = *(const v8bf*)&qeth[g + 8];
      const v8bf w2 = *(const v8bf*)&qetl[g];
      const v8bf w3 = *(const v8bf*)&qetl[g + 8];
      *(v8bf*)&sBh[br * BSTR + bc]     = w0;
      *(v8bf*)&sBh[br * BSTR + bc + 8] = w1;
      *(v8bf*)&sBl[br * BSTR + bc]     = w2;
      *(v8bf*)&sBl[br * BSTR + bc + 8] = w3;
      __syncthreads();
      const int ka = lanelo * ALSTR + k0 + 8 * hi;
      const v16bf Ah = ldfrag(&sALh[ka], &sALh[ka + 16]);
      const v16bf Al = ldfrag(&sALl[ka], &sALl[ka + 16]);
      const int kb = (wave * 16 + lanelo) * BSTR + 8 * hi;
      const v16bf Bh = ldfrag(&sBh[kb], &sBh[kb + 16]);
      const v16bf Bl = ldfrag(&sBl[kb], &sBl[kb + 16]);
      acc = wmma_bf16(Ah, Bh, acc);
      acc = wmma_bf16(Ah, Bl, acc);
      acc = wmma_bf16(Al, Bh, acc);
    }
    const int e = n0 + wave * 16 + lanelo;
#pragma unroll
    for (int r = 0; r < 8; ++r)
      out_align[(b * LD_ + l0 + r + 8 * hi) * E_ + e] = acc[r];
  }
}

// ---------------------------------------------------------------------------
extern "C" void kernel_launch(void* const* d_in, const int* in_sizes, int n_in,
                              void* d_out, int out_size, void* d_ws, size_t ws_size,
                              hipStream_t stream) {
  (void)in_sizes; (void)n_in; (void)out_size; (void)ws_size;
  const float*         d_emb = (const float*)d_in[0];
  const float*         q_emb = (const float*)d_in[1];
  const unsigned char* qmask = (const unsigned char*)d_in[2];  // jnp bool = 1 byte
  const float*         W     = (const float*)d_in[3];
  const float*         bias  = (const float*)d_in[4];

  float* out_align = (float*)d_out;                            // [B, LD, E]
  float* out_alpha = out_align + (size_t)B_ * LD_ * E_;        // [B, LD, LQ]

  // workspace carve (bf16 planes), total ~98.3 MB
  __bf16* wh   = (__bf16*)d_ws;
  __bf16* wl   = wh   + (size_t)E_ * E_;
  __bf16* qph  = wl   + (size_t)E_ * E_;
  __bf16* qpl  = qph  + (size_t)B_ * LQ_ * E_;
  __bf16* qeth = qpl  + (size_t)B_ * LQ_ * E_;
  __bf16* qetl = qeth + (size_t)B_ * E_ * LQ_;

  dim3 blk(256, 1, 1);
  convert_w_kernel<<<dim3((E_ * E_) / 256), blk, 0, stream>>>(W, wh, wl);
  qet_kernel<<<dim3(E_ / 32, LQ_ / 32, B_), blk, 0, stream>>>(q_emb, qeth, qetl);
  qproj_kernel<<<dim3(LQ_ / 16, B_), blk, 70144, stream>>>(q_emb, wh, wl, bias,
                                                           qph, qpl);
  attn_kernel<<<dim3(LD_ / 16, B_), blk, 119808, stream>>>(
      d_emb, q_emb, qmask, bias, wh, wl, qph, qpl, qeth, qetl,
      out_align, out_alpha);
}
